// Informer_73830487818398
// MI455X (gfx1250) — matrix operand order repaired and checked by reading, not compiled, not run on hardware
//
#include <hip/hip_runtime.h>
#include <hip/hip_bf16.h>
#include <math.h>

// ---------------------------------------------------------------------------
// Types for CDNA5 WMMA (wave32), per CDNA5_HIP.md codegen-verified signatures.
// ---------------------------------------------------------------------------
typedef __attribute__((ext_vector_type(16))) _Float16 v16h;
typedef __attribute__((ext_vector_type(8)))  _Float16 v8h;
typedef __attribute__((ext_vector_type(4)))  _Float16 v4h;
typedef __attribute__((ext_vector_type(8)))  float    v8f;
typedef __attribute__((ext_vector_type(4)))  float    v4f;

union U16h { v16h v; v8h h[2]; };

#define D_MODEL 512
#define D_HEAD  64
#define N_HEADS 8

// ---------------------------------------------------------------------------
// Weight convert + transpose: W[K][N] f32 -> Wt[N][K] f16 (done once per call;
// weights total ~39MB f16, stays L2-resident on the 192MB L2).
// ---------------------------------------------------------------------------
__global__ void wcvt_kernel(const float* __restrict__ W, _Float16* __restrict__ Wt,
                            int K, int N)
{
    int o = blockIdx.x * blockDim.x + threadIdx.x;
    if (o >= K * N) return;
    int n = o / K, k = o % K;
    Wt[o] = (_Float16)W[(size_t)k * N + n];
}

// ---------------------------------------------------------------------------
// GEMM: C[M,N] = epilogue(A[M,K] @ W[K,N] + bias), A f32 (LDS-staged -> f16),
// W pre-transposed f16 [N][K] read directly from global in WMMA B layout.
// Block: 256 thr = 8 waves (wm 0..3 x wn 0..1); block tile 128(M) x 64(N);
// each wave owns a 32x32 C tile = 2x2 accumulators = 4 WMMAs per 32-K step.
// act: 0=none, 1=exact GELU (mode 0 only). res: optional residual (mode 0).
// outmode: 0 = f32 row-major, 1 = f16 row-major (for Q/K of attention),
//          2 = f16 per-head transposed Vt[b][h][d][Ls]  (Ls = 1<<lshift).
// ---------------------------------------------------------------------------
__global__ void __launch_bounds__(256)
gemm_kernel(const float* __restrict__ A, const _Float16* __restrict__ Wt,
            const float* __restrict__ bias, const float* __restrict__ res,
            void* __restrict__ Cout, int M, int N, int K, int act,
            int outmode, int lshift)
{
    __shared__ __attribute__((aligned(16))) _Float16 la[128 * 32];  // 8KB

    const int tid  = threadIdx.x;
    const int lane = tid & 31;
    const int w    = tid >> 5;
    const int wm   = w >> 1;       // 0..3
    const int wn   = w & 1;        // 0..1
    const int row0 = blockIdx.y * 128;
    const int col0 = blockIdx.x * 64;
    const int m    = lane & 15;
    const int half = lane >> 4;

    // A staging assignment: 1024 quads (128x32 / 4), 4 per thread.
    const float* pa[4];
    int aoff[4];
#pragma unroll
    for (int j = 0; j < 4; ++j) {
        int q = tid + j * 256;
        int r = q >> 3, c = (q & 7) * 4;
        pa[j]   = A + (size_t)(row0 + r) * K + c;
        aoff[j] = r * 32 + c;
    }
    // B fragment pointers (direct global, f16, row = output column).
    const _Float16* pb0 = Wt + (size_t)(col0 + wn * 32 + m)      * K + half * 16;
    const _Float16* pb1 = Wt + (size_t)(col0 + wn * 32 + 16 + m) * K + half * 16;
    const int arow0 = (wm * 32 + m)      * 32;
    const int arow1 = (wm * 32 + 16 + m) * 32;

    v8f acc[2][2];
#pragma unroll
    for (int i = 0; i < 2; ++i)
#pragma unroll
        for (int j = 0; j < 2; ++j)
#pragma unroll
            for (int r = 0; r < 8; ++r) acc[i][j][r] = 0.f;

    for (int k0 = 0; k0 < K; k0 += 32) {
        // batched global loads first (A f32 quads + B f16 fragments)
        v4f va[4];
#pragma unroll
        for (int j = 0; j < 4; ++j) va[j] = *(const v4f*)pa[j];
        U16h b0, b1;
        b0.h[0] = *(const v8h*)(pb0);
        b0.h[1] = *(const v8h*)(pb0 + 8);
        b1.h[0] = *(const v8h*)(pb1);
        b1.h[1] = *(const v8h*)(pb1 + 8);

        __syncthreads();
#pragma unroll
        for (int j = 0; j < 4; ++j) {
            v4h hh;
#pragma unroll
            for (int t = 0; t < 4; ++t) hh[t] = (_Float16)va[j][t];
            *(v4h*)&la[aoff[j]] = hh;
        }
        __syncthreads();

        U16h a0, a1;
        a0.h[0] = *(const v8h*)&la[arow0 + half * 16];
        a0.h[1] = *(const v8h*)&la[arow0 + half * 16 + 8];
        a1.h[0] = *(const v8h*)&la[arow1 + half * 16];
        a1.h[1] = *(const v8h*)&la[arow1 + half * 16 + 8];

        acc[0][0] = __builtin_amdgcn_wmma_f32_16x16x32_f16(false, a0.v, false, b0.v,
                                                           (short)0, acc[0][0], false, false);
        acc[0][1] = __builtin_amdgcn_wmma_f32_16x16x32_f16(false, a0.v, false, b1.v,
                                                           (short)0, acc[0][1], false, false);
        acc[1][0] = __builtin_amdgcn_wmma_f32_16x16x32_f16(false, a1.v, false, b0.v,
                                                           (short)0, acc[1][0], false, false);
        acc[1][1] = __builtin_amdgcn_wmma_f32_16x16x32_f16(false, a1.v, false, b1.v,
                                                           (short)0, acc[1][1], false, false);
#pragma unroll
        for (int j = 0; j < 4; ++j) pa[j] += 32;
        pb0 += 32;
        pb1 += 32;
    }

    const int Ls = 1 << lshift;
#pragma unroll
    for (int i = 0; i < 2; ++i)
#pragma unroll
        for (int j = 0; j < 2; ++j) {
            int col   = col0 + wn * 32 + j * 16 + m;
            int rbase = row0 + wm * 32 + i * 16 + half * 8;
            float bv  = bias ? bias[col] : 0.f;
#pragma unroll
            for (int r = 0; r < 8; ++r) {
                float v = acc[i][j][r] + bv;
                if (outmode == 0) {
                    if (act == 1) v = 0.5f * v * (1.0f + erff(v * 0.70710678118f));
                    size_t idx = (size_t)(rbase + r) * N + col;
                    if (res) v += res[idx];
                    ((float*)Cout)[idx] = v;
                } else if (outmode == 1) {
                    ((_Float16*)Cout)[(size_t)(rbase + r) * N + col] = (_Float16)v;
                } else {
                    int row  = rbase + r;
                    int bidx = row >> lshift;
                    int l    = row & (Ls - 1);
                    int hh   = col >> 6, d = col & 63;
                    ((_Float16*)Cout)[((size_t)(bidx * N_HEADS + hh) * D_HEAD + d) * Ls + l]
                        = (_Float16)v;
                }
            }
        }
}

// ---------------------------------------------------------------------------
// Flash attention (dense; replaces ProbSparse — sampling/top-k is a latency-
// bound gather pattern; dense WMMA attention is cheaper on MI455X), wave32.
// One wave per 16-query tile per (batch, head). Online softmax over 32-key
// chunks, 8 WMMAs per chunk. All operands read DIRECTLY from global f16 in
// native WMMA layouts (Q/K row-major f16, V pre-transposed per head), so the
// only LDS is the 1KB P-relayout buffer. K/V are L2-resident (<=2MB per b).
// ---------------------------------------------------------------------------
__global__ void __launch_bounds__(32)
flash_attn_kernel(const _Float16* __restrict__ Qh, const _Float16* __restrict__ Kh,
                  const _Float16* __restrict__ Vt, float* __restrict__ O,
                  int Lq, int Lk, int causal)
{
    const int lane = threadIdx.x;
    const int qt   = blockIdx.x;
    const int bh   = blockIdx.y;
    const int b    = bh >> 3;
    const int h    = bh & 7;
    const int hoff = h * D_HEAD;
    const int q0   = qt * 16;
    const int m    = lane & 15, half = lane >> 4;
    const int mbase = half * 8;

    __shared__ __attribute__((aligned(16))) _Float16 lp[16 * 32];

    // Q fragments (A layout) directly from global f16
    U16h aq[2];
    {
        const _Float16* qp = Qh + (size_t)(b * Lq + q0 + m) * D_MODEL + hoff;
#pragma unroll
        for (int c = 0; c < 2; ++c) {
            aq[c].h[0] = *(const v8h*)(qp + c * 32 + half * 16);
            aq[c].h[1] = *(const v8h*)(qp + c * 32 + half * 16 + 8);
        }
    }

    // K fragment pointers: row = key (kb + n), B layout = contiguous 32B
    const _Float16* kp0 = Kh + (size_t)(b * Lk + m) * D_MODEL + hoff + half * 16;
    const _Float16* kp1 = kp0 + (size_t)16 * D_MODEL;
    // Vt fragment base: row = head dim (t*16 + n), cols = keys
    const _Float16* vp  = Vt + ((size_t)(b * N_HEADS + h) * D_HEAD + m) * Lk + half * 16;

    float mrow[8], lrow[8];
    v8f o[4];
#pragma unroll
    for (int r = 0; r < 8; ++r) {
        mrow[r] = -3.0e38f; lrow[r] = 0.f;
        o[0][r] = o[1][r] = o[2][r] = o[3][r] = 0.f;
    }

    const int nKT = Lk / 32;
    for (int kt = 0; kt < nKT; ++kt) {
        const int kb = kt * 32;
        if (causal && kb > q0 + 15) break;
        if (kb + 32 < Lk) {                 // prefetch next chunk (global_prefetch)
            __builtin_prefetch(kp0 + (size_t)32 * D_MODEL, 0, 1);
            __builtin_prefetch(vp + kb + 32, 0, 1);
        }

        // scores: S0 = Q @ K[kb..kb+15]^T, S1 = Q @ K[kb+16..kb+31]^T
        v8f s0, s1;
#pragma unroll
        for (int r = 0; r < 8; ++r) { s0[r] = 0.f; s1[r] = 0.f; }
#pragma unroll
        for (int c = 0; c < 2; ++c) {
            U16h b0, b1;
            b0.h[0] = *(const v8h*)(kp0 + c * 32);
            b0.h[1] = *(const v8h*)(kp0 + c * 32 + 8);
            b1.h[0] = *(const v8h*)(kp1 + c * 32);
            b1.h[1] = *(const v8h*)(kp1 + c * 32 + 8);
            s0 = __builtin_amdgcn_wmma_f32_16x16x32_f16(false, aq[c].v, false, b0.v,
                                                        (short)0, s0, false, false);
            s1 = __builtin_amdgcn_wmma_f32_16x16x32_f16(false, aq[c].v, false, b1.v,
                                                        (short)0, s1, false, false);
        }
#pragma unroll
        for (int r = 0; r < 8; ++r) { s0[r] *= 0.125f; s1[r] *= 0.125f; }
        if (causal) {
#pragma unroll
            for (int r = 0; r < 8; ++r) {
                int qrow = q0 + mbase + r;
                if (kb + m > qrow)      s0[r] = -3.0e38f;
                if (kb + 16 + m > qrow) s1[r] = -3.0e38f;
            }
        }
        // online softmax (row = mbase+r; its 16 cols live in the lane's half)
#pragma unroll
        for (int r = 0; r < 8; ++r) {
            float vmax = fmaxf(s0[r], s1[r]);
            for (int off = 1; off < 16; off <<= 1)
                vmax = fmaxf(vmax, __shfl_xor(vmax, off, 32));
            float mn    = fmaxf(mrow[r], vmax);
            float alpha = __expf(mrow[r] - mn);
            float p0    = __expf(s0[r] - mn);
            float p1    = __expf(s1[r] - mn);
            float rs    = p0 + p1;
            for (int off = 1; off < 16; off <<= 1)
                rs += __shfl_xor(rs, off, 32);
            lrow[r] = lrow[r] * alpha + rs;
            mrow[r] = mn;
            o[0][r] *= alpha; o[1][r] *= alpha; o[2][r] *= alpha; o[3][r] *= alpha;
            lp[(mbase + r) * 32 + m]      = (_Float16)p0;
            lp[(mbase + r) * 32 + 16 + m] = (_Float16)p1;
        }
        __syncthreads();

        // O += P(16x32) @ V(32x64); V fragments direct from Vt rows
        U16h ap;
        ap.h[0] = *(const v8h*)&lp[m * 32 + half * 16];
        ap.h[1] = *(const v8h*)&lp[m * 32 + half * 16 + 8];
#pragma unroll
        for (int t = 0; t < 4; ++t) {
            U16h bvv;
            const _Float16* vrow = vp + (size_t)(t * 16) * Lk + kb;
            bvv.h[0] = *(const v8h*)(vrow);
            bvv.h[1] = *(const v8h*)(vrow + 8);
            o[t] = __builtin_amdgcn_wmma_f32_16x16x32_f16(false, ap.v, false, bvv.v,
                                                          (short)0, o[t], false, false);
        }
        __syncthreads();
        kp0 += (size_t)32 * D_MODEL;
        kp1 += (size_t)32 * D_MODEL;
    }

    float* Ob = O + (size_t)b * Lq * D_MODEL + hoff;
#pragma unroll
    for (int t = 0; t < 4; ++t)
#pragma unroll
        for (int r = 0; r < 8; ++r)
            Ob[(size_t)(q0 + mbase + r) * D_MODEL + t * 16 + m] = o[t][r] / lrow[r];
}

// ---------------------------------------------------------------------------
// Embedding: circular conv1d(k=3, Cin->512, no bias) + sinusoidal PE + time.
// ---------------------------------------------------------------------------
__global__ void embed_kernel(const float* __restrict__ x, const float* __restrict__ xm,
                             const float* __restrict__ tokw, const float* __restrict__ timew,
                             float* __restrict__ out, int Bn, int L, int Cin)
{
    int idx = blockIdx.x * blockDim.x + threadIdx.x;
    int total = Bn * L * D_MODEL;
    if (idx >= total) return;
    int d = idx & (D_MODEL - 1);
    int l = (idx >> 9) % L;
    int b = idx / (D_MODEL * L);

    float acc = 0.f;
    for (int k = 0; k < 3; ++k) {
        int ls = l + k - 1;
        ls = (ls + L) % L;
        const float* xr = x + (size_t)(b * L + ls) * Cin;
        const float* wr = tokw + (size_t)k * Cin * D_MODEL;
        for (int c = 0; c < Cin; ++c) acc += xr[c] * wr[c * D_MODEL + d];
    }
    int d2 = d & ~1;
    float dv  = __expf((float)d2 * (-9.210340371976184f / (float)D_MODEL));
    float ang = (float)l * dv;
    acc += (d & 1) ? __cosf(ang) : __sinf(ang);
    const float* mr = xm + (size_t)(b * L + l) * 4;
    for (int f = 0; f < 4; ++f) acc += mr[f] * timew[f * D_MODEL + d];
    out[idx] = acc;
}

// ---------------------------------------------------------------------------
// LayerNorm over last dim (512). One block (256 threads) per row.
// ---------------------------------------------------------------------------
__global__ void __launch_bounds__(256)
ln_kernel(const float* __restrict__ x, const float* __restrict__ g,
          const float* __restrict__ bb, float* __restrict__ out)
{
    int row = blockIdx.x, tid = threadIdx.x;
    const float* xr = x + (size_t)row * D_MODEL;
    float v0 = xr[2 * tid], v1 = xr[2 * tid + 1];
    float s = v0 + v1, q = v0 * v0 + v1 * v1;
    for (int off = 16; off; off >>= 1) {
        s += __shfl_xor(s, off, 32);
        q += __shfl_xor(q, off, 32);
    }
    __shared__ float as[8], aq[8], mv[2];
    int lane = tid & 31, wv = tid >> 5;
    if (!lane) { as[wv] = s; aq[wv] = q; }
    __syncthreads();
    if (tid == 0) {
        float S = 0.f, Qs = 0.f;
        for (int i = 0; i < 8; ++i) { S += as[i]; Qs += aq[i]; }
        float mm = S / (float)D_MODEL;
        mv[0] = mm;
        mv[1] = rsqrtf(Qs / (float)D_MODEL - mm * mm + 1e-5f);
    }
    __syncthreads();
    float mm = mv[0], rs = mv[1];
    float* orow = out + (size_t)row * D_MODEL;
    orow[2 * tid]     = (v0 - mm) * rs * g[2 * tid]     + bb[2 * tid];
    orow[2 * tid + 1] = (v1 - mm) * rs * g[2 * tid + 1] + bb[2 * tid + 1];
}

// ---------------------------------------------------------------------------
// im2col with circular padding: X[B,L,512] -> out[B*L, 1536]  (k*512 + c)
// ---------------------------------------------------------------------------
__global__ void im2col_kernel(const float* __restrict__ X, float* __restrict__ out,
                              int Bn, int L)
{
    int idx = blockIdx.x * blockDim.x + threadIdx.x;
    int total = Bn * L * 1536;
    if (idx >= total) return;
    int j = idx % 1536;
    int l = (idx / 1536) % L;
    int b = idx / (1536 * L);
    int k = j / D_MODEL, c = j % D_MODEL;
    int ls = (l + k - 1 + L) % L;
    out[idx] = X[(size_t)(b * L + ls) * D_MODEL + c];
}

// ---------------------------------------------------------------------------
// BatchNorm stats (per channel over all rows): st[c]=mean, st[512+c]=var.
// ---------------------------------------------------------------------------
__global__ void __launch_bounds__(256)
bnstats_kernel(const float* __restrict__ y, float* __restrict__ st, int rows)
{
    int c = blockIdx.x, tid = threadIdx.x;
    float s = 0.f, q = 0.f;
    for (int r = tid; r < rows; r += 256) {
        float v = y[(size_t)r * D_MODEL + c];
        s += v; q += v * v;
    }
    for (int off = 16; off; off >>= 1) {
        s += __shfl_xor(s, off, 32);
        q += __shfl_xor(q, off, 32);
    }
    __shared__ float as[8], aq[8];
    int lane = tid & 31, wv = tid >> 5;
    if (!lane) { as[wv] = s; aq[wv] = q; }
    __syncthreads();
    if (tid == 0) {
        float S = 0.f, Q = 0.f;
        for (int i = 0; i < 8; ++i) { S += as[i]; Q += aq[i]; }
        float mm = S / (float)rows;
        st[c] = mm;
        st[D_MODEL + c] = Q / (float)rows - mm * mm;
    }
}

__global__ void bnelu_kernel(float* __restrict__ y, const float* __restrict__ st,
                             const float* __restrict__ g, const float* __restrict__ bb,
                             int rows)
{
    int idx = blockIdx.x * blockDim.x + threadIdx.x;
    int total = rows * D_MODEL;
    if (idx >= total) return;
    int c = idx & (D_MODEL - 1);
    float z = (y[idx] - st[c]) * rsqrtf(st[D_MODEL + c] + 1e-5f) * g[c] + bb[c];
    y[idx] = (z > 0.f) ? z : expm1f(z);
}

// maxpool along L: window 3, stride 2, pad 1. in rows [b*L+l] -> out [b*(L/2)+lo]
__global__ void maxpool_kernel(const float* __restrict__ y, float* __restrict__ out,
                               int Bn, int L)
{
    int Lo = L / 2;
    int idx = blockIdx.x * blockDim.x + threadIdx.x;
    int total = Bn * Lo * D_MODEL;
    if (idx >= total) return;
    int c  = idx & (D_MODEL - 1);
    int lo = (idx >> 9) % Lo;
    int b  = idx / (D_MODEL * Lo);
    float mx = -3.0e38f;
    for (int k = -1; k <= 1; ++k) {
        int li = 2 * lo + k;
        if (li >= 0 && li < L)
            mx = fmaxf(mx, y[(size_t)(b * L + li) * D_MODEL + c]);
    }
    out[idx] = mx;
}

// final projection 512 -> 7, taking last PRED_LEN rows per batch
__global__ void proj_kernel(const float* __restrict__ dec, const float* __restrict__ pw,
                            const float* __restrict__ pb, float* __restrict__ out)
{
    int idx = blockIdx.x * blockDim.x + threadIdx.x;
    int total = 16 * 1024 * 7;
    if (idx >= total) return;
    int c  = idx % 7;
    int lp = (idx / 7) % 1024;
    int b  = idx / (7 * 1024);
    const float* dr = dec + (size_t)(b * 2048 + 1024 + lp) * D_MODEL;
    float acc = pb[c];
    for (int d = 0; d < D_MODEL; ++d) acc += dr[d] * pw[d * 7 + c];
    out[idx] = acc;
}

// ---------------------------------------------------------------------------
// Host orchestration
// ---------------------------------------------------------------------------
struct AP { const float *bk, *bo, *bq, *bv, *wk, *wo, *wq, *wv; };

static inline int ilog2i(int v) { int s = 0; while ((1 << s) < v) ++s; return s; }

extern "C" void kernel_launch(void* const* d_in, const int* in_sizes, int n_in,
                              void* d_out, int out_size, void* d_ws, size_t ws_size,
                              hipStream_t stream)
{
    (void)in_sizes; (void)n_in; (void)out_size; (void)ws_size;
    auto in = [&](int i) { return (const float*)d_in[i]; };

    const float* x_enc      = in(0);
    const float* x_mark_enc = in(1);
    const float* x_dec      = in(2);
    const float* x_mark_dec = in(3);

    // params flattened JAX-pytree style (sorted dict keys, lists in order)
    int base = 4;
    struct ConvP { const float *bn_b, *bn_g, *cb, *cw; } convp[2];
    for (int i = 0; i < 2; ++i) {
        convp[i] = { in(base), in(base + 1), in(base + 2), in(base + 3) };
        base += 4;
    }
    struct DecP {
        const float *c1_b, *c1_w, *c2_b, *c2_w;
        AP cross;
        const float *ln1_b, *ln1_g, *ln2_b, *ln2_g, *ln3_b, *ln3_g;
        AP self_;
    } decp[2];
    for (int i = 0; i < 2; ++i) {
        decp[i].c1_b = in(base + 0); decp[i].c1_w = in(base + 1);
        decp[i].c2_b = in(base + 2); decp[i].c2_w = in(base + 3);
        decp[i].cross = { in(base + 4), in(base + 5), in(base + 6), in(base + 7),
                          in(base + 8), in(base + 9), in(base + 10), in(base + 11) };
        decp[i].ln1_b = in(base + 12); decp[i].ln1_g = in(base + 13);
        decp[i].ln2_b = in(base + 14); decp[i].ln2_g = in(base + 15);
        decp[i].ln3_b = in(base + 16); decp[i].ln3_g = in(base + 17);
        decp[i].self_ = { in(base + 18), in(base + 19), in(base + 20), in(base + 21),
                          in(base + 22), in(base + 23), in(base + 24), in(base + 25) };
        base += 26;
    }
    struct EncP {
        AP a;
        const float *c1_b, *c1_w, *c2_b, *c2_w, *ln1_b, *ln1_g, *ln2_b, *ln2_g;
    } encp[3];
    for (int i = 0; i < 3; ++i) {
        encp[i].a.bk = in(base + 0);  encp[i].a.bo = in(base + 1);
        encp[i].a.bq = in(base + 2);  encp[i].a.bv = in(base + 3);
        encp[i].c1_b = in(base + 4);  encp[i].c1_w = in(base + 5);
        encp[i].c2_b = in(base + 6);  encp[i].c2_w = in(base + 7);
        encp[i].ln1_b = in(base + 8); encp[i].ln1_g = in(base + 9);
        encp[i].ln2_b = in(base + 10); encp[i].ln2_g = in(base + 11);
        encp[i].a.wk = in(base + 12); encp[i].a.wo = in(base + 13);
        encp[i].a.wq = in(base + 14); encp[i].a.wv = in(base + 15);
        base += 16;
    }
    const float* dec_norm_b = in(base++);
    const float* dec_norm_g = in(base++);
    const float* dec_time_w = in(base++);
    const float* dec_tok_w  = in(base++);
    const float* enc_norm_b = in(base++);
    const float* enc_norm_g = in(base++);
    const float* enc_time_w = in(base++);
    const float* enc_tok_w  = in(base++);
    const float* proj_b     = in(base++);
    const float* proj_w     = in(base++);

    // workspace layout (f32 arenas + f16 QKV + f16 weight arena)
    const size_t BIG = (size_t)32768 * 512;      // 16.7M floats
    float* ws  = (float*)d_ws;
    float* X   = ws;
    float* Qf  = X  + BIG;   // reused as f16 Q
    float* Kf  = Qf + BIG;   // reused as f16 K
    float* Vf  = Kf + BIG;   // reused as f16 Vt
    float* Ab  = Vf + BIG;
    float* Tb  = Ab + BIG;
    float* ENC = Tb + BIG;                       // 8192*512
    float* ST  = ENC + (size_t)8192 * 512;       // 1024
    float* Hb  = ST + 1024;                      // 32768*2048
    _Float16* Qh  = (_Float16*)Qf;
    _Float16* Kh  = (_Float16*)Kf;
    _Float16* Vth = (_Float16*)Vf;
    _Float16* wt_cur = (_Float16*)(Hb + (size_t)32768 * 2048);

    auto mkwt = [&](const float* Wsrc, int K, int N) {
        _Float16* dst = wt_cur;
        wt_cur += (size_t)K * N;
        int total = K * N;
        wcvt_kernel<<<(total + 255) / 256, 256, 0, stream>>>(Wsrc, dst, K, N);
        return (const _Float16*)dst;
    };

    // transposed f16 weights
    struct APt { const _Float16 *wq, *wk, *wv, *wo; };
    APt enc_t[3]; const _Float16 *enc_c1t[3], *enc_c2t[3];
    for (int i = 0; i < 3; ++i) {
        enc_t[i].wq = mkwt(encp[i].a.wq, 512, 512);
        enc_t[i].wk = mkwt(encp[i].a.wk, 512, 512);
        enc_t[i].wv = mkwt(encp[i].a.wv, 512, 512);
        enc_t[i].wo = mkwt(encp[i].a.wo, 512, 512);
        enc_c1t[i]  = mkwt(encp[i].c1_w, 512, 2048);
        enc_c2t[i]  = mkwt(encp[i].c2_w, 2048, 512);
    }
    const _Float16* conv_t[2];
    for (int i = 0; i < 2; ++i) conv_t[i] = mkwt(convp[i].cw, 1536, 512);
    APt dec_st[2], dec_ct[2]; const _Float16 *dec_c1t[2], *dec_c2t[2];
    for (int i = 0; i < 2; ++i) {
        dec_st[i].wq = mkwt(decp[i].self_.wq, 512, 512);
        dec_st[i].wk = mkwt(decp[i].self_.wk, 512, 512);
        dec_st[i].wv = mkwt(decp[i].self_.wv, 512, 512);
        dec_st[i].wo = mkwt(decp[i].self_.wo, 512, 512);
        dec_ct[i].wq = mkwt(decp[i].cross.wq, 512, 512);
        dec_ct[i].wk = mkwt(decp[i].cross.wk, 512, 512);
        dec_ct[i].wv = mkwt(decp[i].cross.wv, 512, 512);
        dec_ct[i].wo = mkwt(decp[i].cross.wo, 512, 512);
        dec_c1t[i]   = mkwt(decp[i].c1_w, 512, 2048);
        dec_c2t[i]   = mkwt(decp[i].c2_w, 2048, 512);
    }

    auto gemm = [&](const float* A, const _Float16* Wt, const float* bias,
                    const float* res, float* C, int M, int N, int K, int act) {
        dim3 g(N / 64, M / 128);
        gemm_kernel<<<g, 256, 0, stream>>>(A, Wt, bias, res, (void*)C, M, N, K, act, 0, 0);
    };
    auto gemmh = [&](const float* A, const _Float16* Wt, const float* bias,
                     _Float16* C, int M, int K) {
        dim3 g(512 / 64, M / 128);
        gemm_kernel<<<g, 256, 0, stream>>>(A, Wt, bias, nullptr, (void*)C, M, 512, K, 0, 1, 0);
    };
    auto gemmv = [&](const float* A, const _Float16* Wt, const float* bias,
                     _Float16* C, int M, int K, int lshift) {
        dim3 g(512 / 64, M / 128);
        gemm_kernel<<<g, 256, 0, stream>>>(A, Wt, bias, nullptr, (void*)C, M, 512, K, 0, 2, lshift);
    };
    auto ln = [&](const float* x, const float* g, const float* b, float* o, int rows) {
        ln_kernel<<<rows, 256, 0, stream>>>(x, g, b, o);
    };
    auto attn = [&](const _Float16* q, const _Float16* k, const _Float16* v, float* o,
                    int Lq, int Lk, int causal) {
        dim3 g(Lq / 16, 16 * N_HEADS);
        flash_attn_kernel<<<g, 32, 0, stream>>>(q, k, v, o, Lq, Lk, causal);
    };

    // ---------------- Encoder ----------------
    int L = 2048;
    {
        int total = 16 * L * D_MODEL;
        embed_kernel<<<(total + 255) / 256, 256, 0, stream>>>(
            x_enc, x_mark_enc, enc_tok_w, enc_time_w, X, 16, L, 7);
    }
    for (int i = 0; i < 3; ++i) {
        int M = 16 * L;
        gemmh(X, enc_t[i].wq, encp[i].a.bq, Qh, M, 512);
        gemmh(X, enc_t[i].wk, encp[i].a.bk, Kh, M, 512);
        gemmv(X, enc_t[i].wv, encp[i].a.bv, Vth, M, 512, ilog2i(L));
        attn(Qh, Kh, Vth, Ab, L, L, 0);
        gemm(Ab, enc_t[i].wo, encp[i].a.bo, X, Tb, M, 512, 512, 0);
        ln(Tb, encp[i].ln1_g, encp[i].ln1_b, X, M);
        gemm(X, enc_c1t[i], encp[i].c1_b, nullptr, Hb, M, 2048, 512, 1);
        gemm(Hb, enc_c2t[i], encp[i].c2_b, X, Tb, M, 512, 2048, 0);
        ln(Tb, encp[i].ln2_g, encp[i].ln2_b, X, M);
        if (i < 2) {
            int total = M * 1536;
            im2col_kernel<<<(total + 255) / 256, 256, 0, stream>>>(X, Hb, 16, L);
            gemm(Hb, conv_t[i], convp[i].cb, nullptr, Tb, M, 512, 1536, 0);
            bnstats_kernel<<<512, 256, 0, stream>>>(Tb, ST, M);
            int t2 = M * D_MODEL;
            bnelu_kernel<<<(t2 + 255) / 256, 256, 0, stream>>>(
                Tb, ST, convp[i].bn_g, convp[i].bn_b, M);
            int Lo = L / 2;
            int t3 = 16 * Lo * D_MODEL;
            maxpool_kernel<<<(t3 + 255) / 256, 256, 0, stream>>>(Tb, X, 16, L);
            L = Lo;
        }
    }
    ln(X, enc_norm_g, enc_norm_b, ENC, 16 * L);   // L == 512 here

    // ---------------- Decoder ----------------
    const int Ld = 2048, Md = 16 * Ld, Mk = 16 * 512;
    {
        int total = 16 * Ld * D_MODEL;
        embed_kernel<<<(total + 255) / 256, 256, 0, stream>>>(
            x_dec, x_mark_dec, dec_tok_w, dec_time_w, X, 16, Ld, 7);
    }
    for (int i = 0; i < 2; ++i) {
        // masked self attention
        gemmh(X, dec_st[i].wq, decp[i].self_.bq, Qh, Md, 512);
        gemmh(X, dec_st[i].wk, decp[i].self_.bk, Kh, Md, 512);
        gemmv(X, dec_st[i].wv, decp[i].self_.bv, Vth, Md, 512, 11);
        attn(Qh, Kh, Vth, Ab, Ld, Ld, 1);
        gemm(Ab, dec_st[i].wo, decp[i].self_.bo, X, Tb, Md, 512, 512, 0);
        ln(Tb, decp[i].ln1_g, decp[i].ln1_b, X, Md);
        // cross attention against encoder output (Lk = 512)
        gemmh(X,   dec_ct[i].wq, decp[i].cross.bq, Qh, Md, 512);
        gemmh(ENC, dec_ct[i].wk, decp[i].cross.bk, Kh, Mk, 512);
        gemmv(ENC, dec_ct[i].wv, decp[i].cross.bv, Vth, Mk, 512, 9);
        attn(Qh, Kh, Vth, Ab, Ld, 512, 0);
        gemm(Ab, dec_ct[i].wo, decp[i].cross.bo, X, Tb, Md, 512, 512, 0);
        ln(Tb, decp[i].ln2_g, decp[i].ln2_b, X, Md);
        // FFN
        gemm(X, dec_c1t[i], decp[i].c1_b, nullptr, Hb, Md, 2048, 512, 1);
        gemm(Hb, dec_c2t[i], decp[i].c2_b, X, Tb, Md, 512, 2048, 0);
        ln(Tb, decp[i].ln3_g, decp[i].ln3_b, X, Md);
    }
    ln(X, dec_norm_g, dec_norm_b, Tb, Md);
    {
        int total = 16 * 1024 * 7;
        proj_kernel<<<(total + 255) / 256, 256, 0, stream>>>(
            Tb, proj_w, proj_b, (float*)d_out);
    }
}
